// RouteNet_35132832481982
// MI455X (gfx1250) — matrix operand order repaired and checked by reading, not compiled, not run on hardware
//
#include <hip/hip_runtime.h>
#include <hip/hip_bf16.h>

typedef _Float16 half_t;
typedef __attribute__((ext_vector_type(16))) _Float16 v16h;
typedef __attribute__((ext_vector_type(8)))  _Float16 v8h;
typedef __attribute__((ext_vector_type(8)))  float    v8f;

#define HDIM 512
#define GDIM 1536       // 3*HDIM
#define GEMM_WAVES 4    // waves per workgroup; WG tile = 128(M) x 64(N)

// ---------------------------------------------------------------------------
// WMMA GEMM:  C[M,N] = A[M,K](f16) * W[N,K](f16)^T + bias[N]   (optional ReLU)
// Workgroup = 4 waves; each wave computes a 32x64 sub-tile (2 A frags x 4 B
// frags -> 8 v_wmma_f32_16x16x32_f16 per K-step).  The 4KB B tile per K-step
// is staged cooperatively into LDS once per workgroup and read by all waves
// via ds_load_b128, cutting B-side L2 traffic 4x vs per-wave fetching.
// Requires: M % 128 == 0, N % 64 == 0, K % 32 == 0, buffers 32B aligned.
//
// Lane layouts used by this kernel (wave32; contiguous k order, same for A and B):
//   A (16x32 f16): lane L (r=L&15, hi=L>=16):
//       half i -> A[r][k0+16*hi+i] (contiguous)
//   B (32x16 f16): lane L (n=L&15, hi): half i -> W[n][k0+16*hi+i] (contiguous)
//   C/D (16x16 f32): lane L: vgpr v -> C[v+8*hi][L&15]
// ---------------------------------------------------------------------------
__global__ void gemm_wmma_f16(const half_t* __restrict__ A,
                              const half_t* __restrict__ W,
                              const float* __restrict__ bias,
                              float* __restrict__ C,
                              half_t* __restrict__ C16,   // optional f16 mirror
                              int M, int N, int K, int relu)
{
    __shared__ half_t ldsB[64 * 32];          // [col 0..63][k 0..31], 4 KB

    const int tid  = threadIdx.x;
    const int lane = tid & 31;
    const int wave = tid >> 5;
    const int hi   = lane >> 4;               // 0/1
    const int lr   = lane & 15;
    const int row0 = blockIdx.y * (32 * GEMM_WAVES) + wave * 32;
    const int col0 = blockIdx.x * 64;

    const half_t* A0 = A + (size_t)(row0 + lr) * K + hi * 16;
    const half_t* A1 = A0 + (size_t)16 * K;

    // cooperative B staging: each of 128 threads moves one 32B slice per step
    const int bc = tid >> 1;                  // column 0..63
    const int bp = tid & 1;                   // 16-half part
    const half_t* Wsrc = W + (size_t)(col0 + bc) * K + bp * 16;
    half_t*       Wdst = ldsB + bc * 32 + bp * 16;

    v8f acc[8];
#pragma unroll
    for (int t = 0; t < 4; ++t) {
        float b = bias[col0 + t * 16 + lr];
#pragma unroll
        for (int v = 0; v < 8; ++v) { acc[t][v] = b; acc[4 + t][v] = b; }
    }

    for (int k0 = 0; k0 < K; k0 += 32) {
        __syncthreads();                      // previous tile fully consumed
        *(v16h*)Wdst = *(const v16h*)(Wsrc + k0);
        __syncthreads();                      // tile visible to all waves

        v8h a0lo = *(const v8h*)(A0 + k0);
        v8h a0hi = *(const v8h*)(A0 + k0 + 8);
        v8h a1lo = *(const v8h*)(A1 + k0);
        v8h a1hi = *(const v8h*)(A1 + k0 + 8);
        v16h a0, a1;
#pragma unroll
        for (int i = 0; i < 8; ++i) {
            a0[i] = a0lo[i]; a0[8 + i] = a0hi[i];
            a1[i] = a1lo[i]; a1[8 + i] = a1hi[i];
        }
#pragma unroll
        for (int t = 0; t < 4; ++t) {
            v16h b = *(const v16h*)(ldsB + (t * 16 + lr) * 32 + hi * 16);
            acc[t]     = __builtin_amdgcn_wmma_f32_16x16x32_f16(
                             false, a0, false, b, (short)0, acc[t], false, false);
            acc[4 + t] = __builtin_amdgcn_wmma_f32_16x16x32_f16(
                             false, a1, false, b, (short)0, acc[4 + t], false, false);
        }
    }

#pragma unroll
    for (int h = 0; h < 2; ++h) {
#pragma unroll
        for (int t = 0; t < 4; ++t) {
#pragma unroll
            for (int v = 0; v < 8; ++v) {
                float val = acc[4 * h + t][v];
                if (relu) val = fmaxf(val, 0.0f);
                size_t idx = (size_t)(row0 + 16 * h + v + hi * 8) * N + col0 + t * 16 + lr;
                C[idx] = val;
                if (C16) C16[idx] = (half_t)val;
            }
        }
    }
}

// ---------------------------------------------------------------------------
// GRU gate combine (torch GRUCell math); dual-writes f32 state + f16 mirror.
// ---------------------------------------------------------------------------
__global__ void gru_gates(const float* __restrict__ gi, const float* __restrict__ gh,
                          float* __restrict__ h, half_t* __restrict__ h16, int M)
{
    int idx = blockIdx.x * blockDim.x + threadIdx.x;
    if (idx >= M * HDIM) return;
    int row = idx >> 9;
    int j   = idx & (HDIM - 1);
    const float* gir = gi + (size_t)row * GDIM;
    const float* ghr = gh + (size_t)row * GDIM;
    float i_r = gir[j], i_z = gir[HDIM + j], i_n = gir[2 * HDIM + j];
    float h_r = ghr[j], h_z = ghr[HDIM + j], h_n = ghr[2 * HDIM + j];
    float r = 1.0f / (1.0f + __expf(-(i_r + h_r)));
    float z = 1.0f / (1.0f + __expf(-(i_z + h_z)));
    float n = tanhf(i_n + r * h_n);
    float out = (1.0f - z) * n + z * h[idx];
    h[idx]   = out;
    h16[idx] = (half_t)out;
}

// x16[paths[e],:] = link_h16[links[e],:]  for edges with seqs[e]==s.  block=128
__global__ void build_x16(const int* __restrict__ links, const int* __restrict__ paths,
                          const int* __restrict__ seqs, const half_t* __restrict__ link_h16,
                          half_t* __restrict__ x16, int s)
{
    int e = blockIdx.x;
    if (seqs[e] != s) return;
    const uint2* src = (const uint2*)(link_h16 + (size_t)links[e] * HDIM);
    uint2*       dst = (uint2*)(x16 + (size_t)paths[e] * HDIM);
    dst[threadIdx.x] = src[threadIdx.x];          // 128 x 8B = 512 halves
}

// agg[links[e],:] += path_h[paths[e],:]  (f32 atomics) for edges with seqs[e]==s
__global__ void scatter_add(const int* __restrict__ links, const int* __restrict__ paths,
                            const int* __restrict__ seqs, const float* __restrict__ path_h,
                            float* __restrict__ agg, int s)
{
    int e = blockIdx.x;
    if (seqs[e] != s) return;
    const float* src = path_h + (size_t)paths[e] * HDIM;
    float*       dst = agg + (size_t)links[e] * HDIM;
    for (int j = threadIdx.x; j < HDIM; j += blockDim.x)
        atomicAdd(&dst[j], src[j]);
}

__global__ void zero_f32(float* __restrict__ p, int n)
{
    int i = blockIdx.x * blockDim.x + threadIdx.x;
    if (i < n) p[i] = 0.0f;
}

// h[:,0] = col0 (first Mtrue rows), else 0; dual-writes f16 mirror.
__global__ void init_state(const float* __restrict__ col0, float* __restrict__ h,
                           half_t* __restrict__ h16, int Mpad, int Mtrue)
{
    int i = blockIdx.x * blockDim.x + threadIdx.x;
    if (i >= Mpad * HDIM) return;
    int r = i >> 9, c = i & (HDIM - 1);
    float v = (c == 0 && r < Mtrue) ? col0[r] : 0.0f;
    h[i]   = v;
    h16[i] = (half_t)v;
}

__global__ void cvt_f16(const float* __restrict__ s, half_t* __restrict__ d, int n)
{
    int i = blockIdx.x * blockDim.x + threadIdx.x;
    if (i < n) d[i] = (half_t)s[i];
}

// out[m] = dot(X[m,:Kd], w) + b[0].  One wave per row (wave32 reduce).
__global__ void dense_out(const float* __restrict__ X, const float* __restrict__ w,
                          const float* __restrict__ b, float* __restrict__ out, int Kd)
{
    int row = blockIdx.x;
    const float* xr = X + (size_t)row * Kd;
    float sum = 0.0f;
    for (int j = threadIdx.x; j < Kd; j += 32) sum += xr[j] * w[j];
#pragma unroll
    for (int off = 16; off > 0; off >>= 1) sum += __shfl_down(sum, off, 32);
    if (threadIdx.x == 0) out[row] = sum + b[0];
}

// ---------------------------------------------------------------------------
static inline size_t align256(size_t x) { return (x + 255) & ~(size_t)255; }

extern "C" void kernel_launch(void* const* d_in, const int* in_sizes, int n_in,
                              void* d_out, int out_size, void* d_ws, size_t ws_size,
                              hipStream_t stream)
{
    // ---- inputs (setup_inputs dict order) ----
    const int*   links    = (const int*)d_in[0];
    const int*   paths    = (const int*)d_in[1];
    const int*   seqs     = (const int*)d_in[2];
    const float* link_cap = (const float*)d_in[3];
    const float* bandwdth = (const float*)d_in[4];
    const float* p_Wih = (const float*)d_in[5];
    const float* p_Whh = (const float*)d_in[6];
    const float* p_bih = (const float*)d_in[7];
    const float* p_bhh = (const float*)d_in[8];
    const float* l_Wih = (const float*)d_in[9];
    const float* l_Whh = (const float*)d_in[10];
    const float* l_bih = (const float*)d_in[11];
    const float* l_bhh = (const float*)d_in[12];
    const float* d_w1 = (const float*)d_in[13];
    const float* d_b1 = (const float*)d_in[14];
    const float* d_w2 = (const float*)d_in[15];
    const float* d_b2 = (const float*)d_in[16];
    const float* d_w3 = (const float*)d_in[17];
    const float* d_b3 = (const float*)d_in[18];
    const float* j_w1 = (const float*)d_in[19];
    const float* j_b1 = (const float*)d_in[20];
    const float* j_w2 = (const float*)d_in[21];
    const float* j_b2 = (const float*)d_in[22];
    const float* j_w3 = (const float*)d_in[23];
    const float* j_b3 = (const float*)d_in[24];

    const int E  = in_sizes[0];
    const int NL = in_sizes[3];                 // 3000
    const int NP = in_sizes[4];                 // 4000
    const int NPp = (NP + 127) & ~127;          // 4096 (M tile = 128)
    const int NLp = (NL + 127) & ~127;          // 3072
    const int T = 8, S = 9;

    // ---- workspace carve-out (256B aligned slices) ----
    char* base = (char*)d_ws; size_t off = 0;
    auto take = [&](size_t bytes) { char* p = base + off; off += align256(bytes); return p; };

    float*  path_h   = (float*)take((size_t)NPp * HDIM * 4);
    float*  link_h   = (float*)take((size_t)NLp * HDIM * 4);
    float*  agg      = (float*)take((size_t)NLp * HDIM * 4);
    float*  gi       = (float*)take((size_t)NPp * GDIM * 4);   // shared path/link
    float*  gh       = (float*)take((size_t)NPp * GDIM * 4);
    float*  t1       = (float*)take((size_t)NPp * 256 * 4);
    float*  t2       = (float*)take((size_t)NPp * 128 * 4);
    half_t* path_h16 = (half_t*)take((size_t)NPp * HDIM * 2);
    half_t* link_h16 = (half_t*)take((size_t)NLp * HDIM * 2);
    half_t* x16      = (half_t*)take((size_t)NPp * HDIM * 2);
    half_t* agg16    = (half_t*)take((size_t)NLp * HDIM * 2);
    half_t* t1h      = (half_t*)take((size_t)NPp * 256 * 2);
    half_t* pWih16 = (half_t*)take((size_t)GDIM * HDIM * 2);
    half_t* pWhh16 = (half_t*)take((size_t)GDIM * HDIM * 2);
    half_t* lWih16 = (half_t*)take((size_t)GDIM * HDIM * 2);
    half_t* lWhh16 = (half_t*)take((size_t)GDIM * HDIM * 2);
    half_t* dW116  = (half_t*)take((size_t)256 * HDIM * 2);
    half_t* dW216  = (half_t*)take((size_t)128 * 256 * 2);
    half_t* jW116  = (half_t*)take((size_t)256 * HDIM * 2);
    half_t* jW216  = (half_t*)take((size_t)128 * 256 * 2);
    (void)ws_size; (void)n_in; (void)out_size;

    const int ZB = 256;
    auto grid1 = [&](int n) { return dim3((n + ZB - 1) / ZB); };

    // ---- weight conversion to f16 (every call; deterministic) ----
    cvt_f16<<<grid1(GDIM * HDIM), ZB, 0, stream>>>(p_Wih, pWih16, GDIM * HDIM);
    cvt_f16<<<grid1(GDIM * HDIM), ZB, 0, stream>>>(p_Whh, pWhh16, GDIM * HDIM);
    cvt_f16<<<grid1(GDIM * HDIM), ZB, 0, stream>>>(l_Wih, lWih16, GDIM * HDIM);
    cvt_f16<<<grid1(GDIM * HDIM), ZB, 0, stream>>>(l_Whh, lWhh16, GDIM * HDIM);
    cvt_f16<<<grid1(256 * HDIM), ZB, 0, stream>>>(d_w1, dW116, 256 * HDIM);
    cvt_f16<<<grid1(128 * 256),  ZB, 0, stream>>>(d_w2, dW216, 128 * 256);
    cvt_f16<<<grid1(256 * HDIM), ZB, 0, stream>>>(j_w1, jW116, 256 * HDIM);
    cvt_f16<<<grid1(128 * 256),  ZB, 0, stream>>>(j_w2, jW216, 128 * 256);

    // ---- initial hidden states (f32 + f16 mirror) ----
    init_state<<<grid1(NPp * HDIM), ZB, 0, stream>>>(bandwdth, path_h, path_h16, NPp, NP);
    init_state<<<grid1(NLp * HDIM), ZB, 0, stream>>>(link_cap, link_h, link_h16, NLp, NL);

    // ---- T rounds of message passing ----
    for (int t = 0; t < T; ++t) {
        zero_f32<<<grid1(NLp * HDIM), ZB, 0, stream>>>(agg, NLp * HDIM);

        for (int s = 0; s < S; ++s) {
            // x_s (f16) = scatter(link_h[links]) at rows paths where seqs==s
            zero_f32<<<grid1(NPp * HDIM / 2), ZB, 0, stream>>>((float*)x16, NPp * HDIM / 2);
            build_x16<<<E, 128, 0, stream>>>(links, paths, seqs, link_h16, x16, s);

            // GRU step: gi = x @ Wih^T + bih ; gh = h @ Whh^T + bhh
            gemm_wmma_f16<<<dim3(GDIM / 64, NPp / 128), 128, 0, stream>>>(
                x16, pWih16, p_bih, gi, nullptr, NPp, GDIM, HDIM, 0);
            gemm_wmma_f16<<<dim3(GDIM / 64, NPp / 128), 128, 0, stream>>>(
                path_h16, pWhh16, p_bhh, gh, nullptr, NPp, GDIM, HDIM, 0);
            gru_gates<<<grid1(NPp * HDIM), ZB, 0, stream>>>(gi, gh, path_h, path_h16, NPp);

            // seq_out[s] == path_h now: fold gather + segment-sum into agg
            scatter_add<<<E, 128, 0, stream>>>(links, paths, seqs, path_h, agg, s);
        }

        // link GRU: link_h = GRU(agg, link_h)
        cvt_f16<<<grid1(NLp * HDIM), ZB, 0, stream>>>(agg, agg16, NLp * HDIM);
        gemm_wmma_f16<<<dim3(GDIM / 64, NLp / 128), 128, 0, stream>>>(
            agg16, lWih16, l_bih, gi, nullptr, NLp, GDIM, HDIM, 0);
        gemm_wmma_f16<<<dim3(GDIM / 64, NLp / 128), 128, 0, stream>>>(
            link_h16, lWhh16, l_bhh, gh, nullptr, NLp, GDIM, HDIM, 0);
        gru_gates<<<grid1(NLp * HDIM), ZB, 0, stream>>>(gi, gh, link_h, link_h16, NLp);
    }

    // ---- readout MLPs (d then j), output concat into d_out ----
    float* out_d = (float*)d_out;
    float* out_j = (float*)d_out + NP;

    gemm_wmma_f16<<<dim3(256 / 64, NPp / 128), 128, 0, stream>>>(
        path_h16, dW116, d_b1, t1, t1h, NPp, 256, HDIM, 1);
    gemm_wmma_f16<<<dim3(128 / 64, NPp / 128), 128, 0, stream>>>(
        t1h, dW216, d_b2, t2, nullptr, NPp, 128, 256, 1);
    dense_out<<<NP, 32, 0, stream>>>(t2, d_w3, d_b3, out_d, 128);

    gemm_wmma_f16<<<dim3(256 / 64, NPp / 128), 128, 0, stream>>>(
        path_h16, jW116, j_b1, t1, t1h, NPp, 256, HDIM, 1);
    gemm_wmma_f16<<<dim3(128 / 64, NPp / 128), 128, 0, stream>>>(
        t1h, jW216, j_b2, t2, nullptr, NPp, 128, 256, 1);
    dense_out<<<NP, 32, 0, stream>>>(t2, j_w3, j_b3, out_j, 128);
}
